// TransformerMultiInputBlock_22428319220358
// MI455X (gfx1250) — matrix operand order repaired and checked by reading, not compiled, not run on hardware
//
#include <hip/hip_runtime.h>
#include <hip/hip_bf16.h>

// ---------------------------------------------------------------------------
// MI455X (gfx1250) transformer multi-input block, bf16 WMMA everywhere.
// ---------------------------------------------------------------------------

typedef __bf16 bf16_t;
typedef __attribute__((ext_vector_type(16))) __bf16 v16bf;
typedef __attribute__((ext_vector_type(8)))  float  v8f;
typedef unsigned int u32;

#define BB     16
#define SS     512
#define HH     512
#define NHEAD  8
#define HSZ    64
#define DFFN   2048
#define MROWS  (BB*SS)          /* 8192 token rows */
#define NEGV   (-1.0e9f)
#define EPSL   (1e-6f)

#ifndef __has_builtin
#define __has_builtin(x) 0
#endif

// Async memory->LDS staging is only emitted in device compilation; the host
// pass parses the synchronous fallback.
#if defined(__HIP_DEVICE_COMPILE__)
#define HAVE_ASYNC_COPY 1
#else
#define HAVE_ASYNC_COPY 0
#endif

union Frag16 { v16bf v; u32 u[8]; };

static __device__ inline v8f vzero8() {
    v8f z;
    for (int i = 0; i < 8; ++i) z[i] = 0.0f;
    return z;
}

static __device__ __forceinline__ void async_wait0()
{
#if defined(__HIP_DEVICE_COMPILE__)
#if __has_builtin(__builtin_amdgcn_s_wait_asynccnt)
    __builtin_amdgcn_s_wait_asynccnt(0);
#else
    asm volatile("s_wait_asynccnt 0x0" ::: "memory");
#endif
#endif
}

#if HAVE_ASYNC_COPY
// GLOBAL_LOAD_ASYNC_TO_LDS_B128 (VFLAT op 98, ASYNCcnt-tracked): vdst holds the
// LDS byte offset, vaddr the 64-bit global address.  The clang builtin's
// parameters use the unspellable __device__/__shared__ language address
// spaces, so issue the instruction via inline asm instead.
static __device__ __forceinline__ void async_copy16(const void* g, void* lds)
{
    unsigned int loff = (unsigned int)(uintptr_t)lds;  // low 32 bits = LDS offset
    asm volatile("global_load_async_to_lds_b128 %0, %1, off"
                 :
                 : "v"(loff), "v"(g)
                 : "memory");
}
#endif

// ---------------------------------------------------------------------------
// Generic bf16 WMMA GEMM:  C[M,N] = A[M,K] * W[K,N] + bias  (W given as Bt[N][K])
// Block: 128 threads (4 wave32).  Tile: 128(M) x 64(N), K-step 32.
// Each wave: 32(M) x 64(N) -> 2 A-frags + 4 B-frags feed 8 WMMAs per K-step.
// MODE 0: row-major fp32 (outF) and/or bf16 (outB)
// MODE 1: bf16 out in Q/K head layout   [b, h, s, d]
// MODE 2: bf16 out in V^T head layout   [b, h, d, s]
// ---------------------------------------------------------------------------
template<int MODE, int RELU>
__global__ __launch_bounds__(128)
void gemm_bf16_kernel(const bf16_t* __restrict__ A,
                      const bf16_t* __restrict__ Bt,
                      const float*  __restrict__ bias,
                      float*        __restrict__ outF,
                      bf16_t*       __restrict__ outB,
                      int N, int K)
{
    __shared__ __align__(16) bf16_t As[128][40];
    __shared__ __align__(16) bf16_t Bs[64][40];

    const int tid  = threadIdx.x;
    const int lane = tid & 31;
    const int wave = tid >> 5;
    const int m0   = blockIdx.x * 128;
    const int n0   = blockIdx.y * 64;
    const int lr   = lane & 15;      // N / M-within-16 index
    const int hi   = lane >> 4;      // half-select

    v8f acc[2][4];
    for (int hf = 0; hf < 2; ++hf)
        for (int t = 0; t < 4; ++t) acc[hf][t] = vzero8();

    const int brow = tid >> 1, bhalf = tid & 1;   // Bt tile: 64 rows x 32 cols

    for (int kb = 0; kb < K; kb += 32) {
        // Stage A (128x32) and Bt (64x32) tiles into LDS.
#if HAVE_ASYNC_COPY
        // CDNA5 async memory->LDS path (no VGPR round-trip), tracked by ASYNCcnt.
        for (int q = 0; q < 4; ++q)
            async_copy16(&A[(size_t)(m0 + tid) * K + kb + q * 8], &As[tid][q * 8]);
        for (int q = 0; q < 2; ++q)
            async_copy16(&Bt[(size_t)(n0 + brow) * K + kb + bhalf * 16 + q * 8],
                         &Bs[brow][bhalf * 16 + q * 8]);
        async_wait0();
#else
        for (int q = 0; q < 4; ++q)
            *(uint4*)&As[tid][q * 8] =
                *(const uint4*)&A[(size_t)(m0 + tid) * K + kb + q * 8];
        for (int q = 0; q < 2; ++q)
            *(uint4*)&Bs[brow][bhalf * 16 + q * 8] =
                *(const uint4*)&Bt[(size_t)(n0 + brow) * K + kb + bhalf * 16 + q * 8];
        if (kb + 32 < K) {
            __builtin_prefetch(&A[(size_t)(m0 + tid) * K + kb + 32], 0, 1);
            __builtin_prefetch(&Bt[(size_t)(n0 + brow) * K + kb + 32 + bhalf * 16], 0, 1);
        }
#endif
        __syncthreads();

        // Two A fragments (rows wave*32 + {0..15} and + {16..31}):
        // ISA A layout: K = (j%4)*2 + (j/4)*16 + hi*8 (+h), M = lr
        Frag16 af0, af1;
        const int ml0 = wave * 32 + lr;
        const int ml1 = wave * 32 + 16 + lr;
        for (int j = 0; j < 8; ++j) {
            int k0 = ((j & 3) * 2) + ((j >> 2) * 16) + hi * 8;
            af0.u[j] = *(const u32*)&As[ml0][k0];
            af1.u[j] = *(const u32*)&As[ml1][k0];
        }
        // 4 x B fragment (32x16): K = (j%4)*2 + (j/4)*16 + hi*8 (+h) ; N = lr
        for (int t = 0; t < 4; ++t) {
            Frag16 bfr;
            int nl = t * 16 + lr;
            for (int j = 0; j < 8; ++j)
                bfr.u[j] = *(const u32*)&Bs[nl][((j & 3) * 2) + ((j >> 2) * 16) + hi * 8];
            acc[0][t] = __builtin_amdgcn_wmma_f32_16x16x32_bf16(
                false, af0.v, false, bfr.v, (short)0, acc[0][t], false, false);
            acc[1][t] = __builtin_amdgcn_wmma_f32_16x16x32_bf16(
                false, af1.v, false, bfr.v, (short)0, acc[1][t], false, false);
        }
        __syncthreads();
    }

    // Epilogue: C layout row = hi*8 + r, col = lr
    for (int hf = 0; hf < 2; ++hf)
        for (int t = 0; t < 4; ++t) {
            int n = n0 + t * 16 + lr;
            float bi = bias[n];
            for (int r = 0; r < 8; ++r) {
                int m = m0 + wave * 32 + hf * 16 + hi * 8 + r;
                float v = acc[hf][t][r] + bi;
                if (RELU) v = fmaxf(v, 0.0f);
                if (MODE == 0) {
                    if (outF) outF[(size_t)m * N + n] = v;
                    if (outB) outB[(size_t)m * N + n] = (bf16_t)v;
                } else if (MODE == 1) {
                    int b = m >> 9, s = m & 511, h = n >> 6, d = n & 63;
                    outB[(((size_t)(b * NHEAD + h)) * SS + s) * HSZ + d] = (bf16_t)v;
                } else {
                    int b = m >> 9, s = m & 511, h = n >> 6, d = n & 63;
                    outB[(((size_t)(b * NHEAD + h)) * HSZ + d) * SS + s] = (bf16_t)v;
                }
            }
        }
}

// ---------------------------------------------------------------------------
// Fused causal flash attention.  One wave32 per (b, h, 16-query tile).
// QH/KH: [b,h,s,d] bf16.  VT: [b,h,d,s] bf16.  Output: [b*S, H] bf16 (heads concat).
// ---------------------------------------------------------------------------
__global__ __launch_bounds__(32)
void attn_kernel(const bf16_t* __restrict__ QH,
                 const bf16_t* __restrict__ KH,
                 const bf16_t* __restrict__ VT,
                 const int*    __restrict__ act,
                 bf16_t*       __restrict__ attnB)
{
    __shared__ __align__(16) bf16_t pt[16][32];

    const int lane  = threadIdx.x;
    const int qbase = blockIdx.x * 16;
    const int bh    = blockIdx.y;
    const int b     = bh >> 3;
    const int h     = bh & 7;
    const int lr    = lane & 15;
    const int hi    = lane >> 4;

    const bf16_t* Q  = QH + (size_t)bh * SS * HSZ;
    const bf16_t* Kp = KH + (size_t)bh * SS * HSZ;
    const bf16_t* Vp = VT + (size_t)bh * HSZ * SS;

    // Q fragments (rows qbase..+15): d halves [0,32) and [32,64)
    Frag16 qa0, qa1;
    for (int j = 0; j < 8; ++j) {
        int d0 = ((j & 3) * 2) + ((j >> 2) * 16) + hi * 8;
        qa0.u[j] = *(const u32*)&Q[(size_t)(qbase + lr) * HSZ + d0];
        qa1.u[j] = *(const u32*)&Q[(size_t)(qbase + lr) * HSZ + 32 + d0];
    }

    v8f o[4];
    for (int t = 0; t < 4; ++t) o[t] = vzero8();
    float mi[8], li[8];
    int dead[8];
    for (int r = 0; r < 8; ++r) {
        mi[r] = -1.0e30f; li[r] = 0.0f;
        dead[r] = (act[b * SS + qbase + hi * 8 + r] == 0);
    }

    const int ntiles = (qbase + 15) / 32 + 1;     // causal limit, 32 keys per tile
    for (int kt = 0; kt < ntiles; ++kt) {
        const int k0 = kt * 32;

        // K fragments: [sub(16-key group)][d-half]
        Frag16 kfr[4];
        for (int sub = 0; sub < 2; ++sub)
            for (int dh = 0; dh < 2; ++dh)
                for (int j = 0; j < 8; ++j) {
                    int key = k0 + sub * 16 + lr;
                    int dd  = dh * 32 + ((j & 3) * 2) + ((j >> 2) * 16) + hi * 8;
                    kfr[sub * 2 + dh].u[j] = *(const u32*)&Kp[(size_t)key * HSZ + dd];
                }

        v8f c0 = vzero8(), c1 = vzero8();
        c0 = __builtin_amdgcn_wmma_f32_16x16x32_bf16(false, qa0.v, false, kfr[0].v, (short)0, c0, false, false);
        c0 = __builtin_amdgcn_wmma_f32_16x16x32_bf16(false, qa1.v, false, kfr[1].v, (short)0, c0, false, false);
        c1 = __builtin_amdgcn_wmma_f32_16x16x32_bf16(false, qa0.v, false, kfr[2].v, (short)0, c1, false, false);
        c1 = __builtin_amdgcn_wmma_f32_16x16x32_bf16(false, qa1.v, false, kfr[3].v, (short)0, c1, false, false);

        // online softmax per row (rows striped hi*8+r, cols = lanes within half)
        for (int r = 0; r < 8; ++r) {
            int qrow = qbase + hi * 8 + r;
            float s0 = c0[r] * 0.125f;
            float s1 = c1[r] * 0.125f;
            if ((k0 + lr)      > qrow || dead[r]) s0 = NEGV;
            if ((k0 + 16 + lr) > qrow || dead[r]) s1 = NEGV;
            float t = fmaxf(s0, s1);
            t = fmaxf(t, __shfl_xor(t, 1, 32));
            t = fmaxf(t, __shfl_xor(t, 2, 32));
            t = fmaxf(t, __shfl_xor(t, 4, 32));
            t = fmaxf(t, __shfl_xor(t, 8, 32));
            float mn    = fmaxf(mi[r], t);
            float alpha = __expf(mi[r] - mn);
            float p0    = __expf(s0 - mn);
            float p1    = __expf(s1 - mn);
            float rs = p0 + p1;
            rs += __shfl_xor(rs, 1, 32);
            rs += __shfl_xor(rs, 2, 32);
            rs += __shfl_xor(rs, 4, 32);
            rs += __shfl_xor(rs, 8, 32);
            li[r] = li[r] * alpha + rs;
            mi[r] = mn;
            for (int t4 = 0; t4 < 4; ++t4) o[t4][r] *= alpha;
            int rl = hi * 8 + r;
            pt[rl][lr]      = (bf16_t)p0;
            pt[rl][16 + lr] = (bf16_t)p1;
        }
        __syncthreads();

        // P fragment (16 rows x 32 keys) from LDS, then P*V per 16-wide d tile
        Frag16 pa;
        for (int j = 0; j < 8; ++j) {
            int kk = ((j & 3) * 2) + ((j >> 2) * 16) + hi * 8;
            pa.u[j] = *(const u32*)&pt[lr][kk];
        }
        for (int t4 = 0; t4 < 4; ++t4) {
            Frag16 vb;
            int dcol = t4 * 16 + lr;
            for (int j = 0; j < 8; ++j) {
                int kk = k0 + ((j & 3) * 2) + ((j >> 2) * 16) + hi * 8;
                vb.u[j] = *(const u32*)&Vp[(size_t)dcol * SS + kk];
            }
            o[t4] = __builtin_amdgcn_wmma_f32_16x16x32_bf16(
                false, pa.v, false, vb.v, (short)0, o[t4], false, false);
        }
        __syncthreads();
    }

    // normalize and write heads-concatenated bf16 [b*S, H]
    for (int t4 = 0; t4 < 4; ++t4)
        for (int r = 0; r < 8; ++r) {
            int rl = hi * 8 + r;
            float v = o[t4][r] / li[r];
            attnB[((size_t)(b * SS + qbase + rl)) * HH + h * HSZ + t4 * 16 + lr] = (bf16_t)v;
        }
}

// ---------------------------------------------------------------------------
// LayerNorm(x + res) matching reference (unbiased std, eps added to std).
// One wave per 512-wide row.
// ---------------------------------------------------------------------------
__global__ __launch_bounds__(256)
void ln_kernel(const float* __restrict__ x, const float* __restrict__ res,
               const float* __restrict__ w, const float* __restrict__ bcoef,
               float* __restrict__ yF, bf16_t* __restrict__ yB)
{
    const int wave = threadIdx.x >> 5;
    const int lane = threadIdx.x & 31;
    const int row  = blockIdx.x * 8 + wave;

    const float* xr = x   + (size_t)row * HH;
    const float* rr = res + (size_t)row * HH;
    float vals[16];
    float sum = 0.0f;
    for (int i = 0; i < 16; ++i) {
        float v = xr[lane + i * 32] + rr[lane + i * 32];
        vals[i] = v; sum += v;
    }
    sum += __shfl_xor(sum, 1, 32);  sum += __shfl_xor(sum, 2, 32);
    sum += __shfl_xor(sum, 4, 32);  sum += __shfl_xor(sum, 8, 32);
    sum += __shfl_xor(sum, 16, 32);
    float mean = sum * (1.0f / 512.0f);
    float sq = 0.0f;
    for (int i = 0; i < 16; ++i) { float d = vals[i] - mean; sq += d * d; }
    sq += __shfl_xor(sq, 1, 32);  sq += __shfl_xor(sq, 2, 32);
    sq += __shfl_xor(sq, 4, 32);  sq += __shfl_xor(sq, 8, 32);
    sq += __shfl_xor(sq, 16, 32);
    float istd = 1.0f / (sqrtf(sq * (1.0f / 511.0f)) + EPSL);
    for (int i = 0; i < 16; ++i) {
        int c = lane + i * 32;
        float y = w[c] * (vals[i] - mean) * istd + bcoef[c];
        yF[(size_t)row * HH + c] = y;
        if (yB) yB[(size_t)row * HH + c] = (bf16_t)y;
    }
}

// ---------------------------------------------------------------------------
// Elementwise helpers
// ---------------------------------------------------------------------------
__global__ __launch_bounds__(256)
void cvt_kernel(const float* __restrict__ in, bf16_t* __restrict__ out, int n)
{
    int i = blockIdx.x * 256 + threadIdx.x;
    if (i < n) out[i] = (bf16_t)in[i];
}

__global__ __launch_bounds__(256)
void add2_kernel(const float* __restrict__ a, const float* __restrict__ b,
                 float* __restrict__ outF, bf16_t* __restrict__ outB, int n)
{
    int i = blockIdx.x * 256 + threadIdx.x;
    if (i < n) { float v = a[i] + b[i]; outF[i] = v; outB[i] = (bf16_t)v; }
}

__global__ __launch_bounds__(256)
void wtrans_kernel(const float* __restrict__ W, bf16_t* __restrict__ Bt, int N, int K)
{
    int i = blockIdx.x * 256 + threadIdx.x;
    if (i >= N * K) return;
    int n = i / K, k = i - n * K;
    Bt[i] = (bf16_t)W[(size_t)k * N + n];
}

// ---------------------------------------------------------------------------
// Host orchestration
// ---------------------------------------------------------------------------
static inline void* bump(char*& p, size_t sz)
{
    void* r = (void*)p;
    p += (sz + 255) & ~(size_t)255;
    return r;
}

extern "C" void kernel_launch(void* const* d_in, const int* in_sizes, int n_in,
                              void* d_out, int out_size, void* d_ws, size_t ws_size,
                              hipStream_t stream)
{
    (void)in_sizes; (void)n_in; (void)out_size; (void)ws_size;

    const float* x_t = (const float*)d_in[0];
    const float* x_o = (const float*)d_in[1];
    const float* x_s = (const float*)d_in[2];
    const int*   act = (const int*)  d_in[3];
    // params: sat(4..13) sao(14..23) cto(24..33) cot(34..43) ff_t(44..49) ff_o(50..55)

    const size_t SZ_F   = (size_t)MROWS * HH * sizeof(float);   // 16 MB
    const size_t SZ_B   = (size_t)MROWS * HH * sizeof(bf16_t);  //  8 MB
    const size_t SZ_W   = (size_t)HH * HH * sizeof(bf16_t);     // 512 KB
    const size_t SZ_WFF = (size_t)HH * DFFN * sizeof(bf16_t);   //  2 MB

    char* p = (char*)d_ws;
    bf16_t* wt[4][4];
    for (int mi = 0; mi < 4; ++mi)
        for (int wi = 0; wi < 4; ++wi)
            wt[mi][wi] = (bf16_t*)bump(p, SZ_W);
    bf16_t* w1t_t = (bf16_t*)bump(p, SZ_WFF);
    bf16_t* w2t_t = (bf16_t*)bump(p, SZ_WFF);
    bf16_t* w1t_o = (bf16_t*)bump(p, SZ_WFF);
    bf16_t* w2t_o = (bf16_t*)bump(p, SZ_WFF);
    bf16_t* xtB   = (bf16_t*)bump(p, SZ_B);
    bf16_t* xoB   = (bf16_t*)bump(p, SZ_B);
    char*   qkv0  = p;                          // QH+KH+VT+ATTN region (32 MB),
    bf16_t* QHb   = (bf16_t*)bump(p, SZ_B);     // aliased by HID during FFN phase
    bf16_t* KHb   = (bf16_t*)bump(p, SZ_B);
    bf16_t* VTb   = (bf16_t*)bump(p, SZ_B);
    bf16_t* ATTNb = (bf16_t*)bump(p, SZ_B);
    bf16_t* HID   = (bf16_t*)qkv0;
    float*  PROJ  = (float*)bump(p, SZ_F);
    float*  S1    = (float*)bump(p, SZ_F);      // x_t_
    bf16_t* S1B   = (bf16_t*)bump(p, SZ_B);
    float*  S2    = (float*)bump(p, SZ_F);      // x_to_
    float*  S3    = (float*)bump(p, SZ_F);      // x_o_
    bf16_t* S3B   = (bf16_t*)bump(p, SZ_B);
    float*  S4    = (float*)bump(p, SZ_F);      // x_ot_
    float*  FIN   = (float*)bump(p, SZ_F);
    bf16_t* FINB  = (bf16_t*)bump(p, SZ_B);

    float* out_t = (float*)d_out;
    float* out_o = out_t + (size_t)MROWS * HH;

    const int NELT = MROWS * HH;
    const dim3 blk256(256), grELT((NELT + 255) / 256);
    const dim3 blkG(128);
    const dim3 grG512(MROWS / 128, HH / 64);
    const dim3 grGDFF(MROWS / 128, DFFN / 64);
    const dim3 grATT(SS / 16, BB * NHEAD), blkATT(32);
    const dim3 grLN(MROWS / 8);

    // --- weight transpose + bf16 conversion --------------------------------
    for (int mi = 0; mi < 4; ++mi) {
        int pb = 4 + mi * 10;
        for (int wi = 0; wi < 4; ++wi) {
            const float* W = (const float*)d_in[pb + wi * 2];  // wq,wk,wv,wo
            wtrans_kernel<<<(HH * HH + 255) / 256, blk256, 0, stream>>>(W, wt[mi][wi], HH, HH);
        }
    }
    wtrans_kernel<<<(HH * DFFN + 255) / 256, blk256, 0, stream>>>((const float*)d_in[44], w1t_t, DFFN, HH);
    wtrans_kernel<<<(HH * DFFN + 255) / 256, blk256, 0, stream>>>((const float*)d_in[46], w2t_t, HH, DFFN);
    wtrans_kernel<<<(HH * DFFN + 255) / 256, blk256, 0, stream>>>((const float*)d_in[50], w1t_o, DFFN, HH);
    wtrans_kernel<<<(HH * DFFN + 255) / 256, blk256, 0, stream>>>((const float*)d_in[52], w2t_o, HH, DFFN);

    cvt_kernel<<<grELT, blk256, 0, stream>>>(x_t, xtB, NELT);
    cvt_kernel<<<grELT, blk256, 0, stream>>>(x_o, xoB, NELT);

    // --- one MHA block -----------------------------------------------------
    auto run_mha = [&](int mi, const float* qF, const bf16_t* qB,
                       const bf16_t* kB, const bf16_t* vB,
                       float* outF, bf16_t* outB) {
        int pb = 4 + mi * 10;
        gemm_bf16_kernel<1, 0><<<grG512, blkG, 0, stream>>>(
            qB, wt[mi][0], (const float*)d_in[pb + 1], nullptr, QHb, HH, HH);
        gemm_bf16_kernel<1, 0><<<grG512, blkG, 0, stream>>>(
            kB, wt[mi][1], (const float*)d_in[pb + 3], nullptr, KHb, HH, HH);
        gemm_bf16_kernel<2, 0><<<grG512, blkG, 0, stream>>>(
            vB, wt[mi][2], (const float*)d_in[pb + 5], nullptr, VTb, HH, HH);
        attn_kernel<<<grATT, blkATT, 0, stream>>>(QHb, KHb, VTb, act, ATTNb);
        gemm_bf16_kernel<0, 0><<<grG512, blkG, 0, stream>>>(
            ATTNb, wt[mi][3], (const float*)d_in[pb + 7], PROJ, nullptr, HH, HH);
        ln_kernel<<<grLN, blk256, 0, stream>>>(
            PROJ, qF, (const float*)d_in[pb + 8], (const float*)d_in[pb + 9], outF, outB);
    };

    run_mha(0, x_t, xtB, xtB, xtB, S1, S1B);     // sat : x_t_
    run_mha(2, S1,  S1B, xoB, xoB, S2, nullptr); // cto : x_to_
    run_mha(1, x_o, xoB, xoB, xoB, S3, S3B);     // sao : x_o_
    run_mha(3, S3,  S3B, S1B, S1B, S4, nullptr); // cot : x_ot_

    // --- one FFN block -----------------------------------------------------
    auto run_ffn = [&](const float* srcF, const bf16_t* w1t, const bf16_t* w2t,
                       int pb, float* outF) {
        add2_kernel<<<grELT, blk256, 0, stream>>>(srcF, x_s, FIN, FINB, NELT);
        gemm_bf16_kernel<0, 1><<<grGDFF, blkG, 0, stream>>>(
            FINB, w1t, (const float*)d_in[pb + 1], nullptr, HID, DFFN, HH);
        gemm_bf16_kernel<0, 0><<<grG512, blkG, 0, stream>>>(
            HID, w2t, (const float*)d_in[pb + 3], PROJ, nullptr, HH, DFFN);
        ln_kernel<<<grLN, blk256, 0, stream>>>(
            PROJ, FIN, (const float*)d_in[pb + 4], (const float*)d_in[pb + 5], outF, nullptr);
    };

    run_ffn(S2, w1t_t, w2t_t, 44, out_t);        // ff_t
    run_ffn(S4, w1t_o, w2t_o, 50, out_o);        // ff_o
}
